// SPELL_25555055412071
// MI455X (gfx1250) — compile-verified, matrix operand order checked by reading, not run on hardware
//
#include <hip/hip_runtime.h>

// ---------------------------------------------------------------------------
// MI455X (gfx1250, wave32) implementation.
//  * EdgeConv layer-1 factored into node GEMMs (linearity); layer-2 is a
//    per-edge 16x64 @ 64x64 WMMA GEMM staged via LDS; max-agg via ordered-uint
//    atomicMax.
//  * RGCN: scatter-add first, then fused K=192 WMMA GEMM (aggregate-then-GEMM).
//  * Refinement: all channel mixes via v_wmma_f32_16x16x32_f16; dilated conv =
//    3 shifted-row GEMMs with zero-padded fragments.
//  * GEMM/conv waves process TWO 16-row tiles so each weight fragment load
//    feeds two WMMAs (halves B-matrix vmem traffic, doubles wmma density).
// ---------------------------------------------------------------------------

typedef _Float16 h16;
typedef __attribute__((ext_vector_type(8)))  _Float16 v8h;
typedef __attribute__((ext_vector_type(16))) _Float16 v16h;
typedef __attribute__((ext_vector_type(8)))  float    v8f;
typedef __attribute__((ext_vector_type(4)))  float    v4f;

// ---- WMMA fragment helpers (CDNA5 16-bit A/B layout, wave32) --------------
// A 16x32: lane L (row = L&15, hl = L>>4) holds K = {hl*8..hl*8+7, 16+hl*8..+7}
// B 32x16 mirrored: per-lane column n = L&15 from Wt[n][k] (k contiguous).
__device__ __forceinline__ v16h frag_load(const h16* rowp, int hl) {
  v8h lo = *(const v8h*)(rowp + hl * 8);
  v8h hi = *(const v8h*)(rowp + 16 + hl * 8);
  v16h r;
#pragma unroll
  for (int i = 0; i < 8; i++) { r[i] = lo[i]; r[8 + i] = hi[i]; }
  return r;
}

__device__ __forceinline__ v16h frag_zero() {
  v16h r;
#pragma unroll
  for (int i = 0; i < 16; i++) r[i] = (_Float16)0.0f;
  return r;
}

// ordered-uint encoding for float atomic max
__device__ __forceinline__ unsigned ordF(float f) {
  unsigned u = __float_as_uint(f);
  return (u & 0x80000000u) ? ~u : (u | 0x80000000u);
}
__device__ __forceinline__ float decU(unsigned u) {
  if (u == 0u) return 0.0f;                       // no writer -> PyG fill 0
  unsigned v = (u & 0x80000000u) ? (u & 0x7fffffffu) : ~u;
  if ((v & 0x7f800000u) == 0x7f800000u) return 0.0f;  // non-finite -> 0
  return __uint_as_float(v);
}

// ---------------------------------------------------------------------------
// Generic WMMA GEMM: C[M,64] (+)= A[M,K] @ Wt^T (+ bias), Wt stored [64][K] f16.
// One wave = 32 rows (two 16-row tiles) x 64 cols; each B fragment feeds 2 WMMAs.
// ---------------------------------------------------------------------------
template <int K, bool ACCUM, bool OUTH>
__global__ void gemm16(const h16* __restrict__ A, int lda,
                       const h16* __restrict__ Wt,
                       const float* __restrict__ bias,
                       void* __restrict__ Cout, int M) {
  int wave = threadIdx.x >> 5, lane = threadIdx.x & 31;
  int r0 = (blockIdx.x * 8 + wave) * 32;
  if (r0 >= M) return;                    // whole-wave uniform exit, no barriers
  int rl = lane & 15, hl = lane >> 4;

  v8f acc[2][4];
#pragma unroll
  for (int t = 0; t < 4; t++) {
    float bv = bias ? bias[t * 16 + rl] : 0.0f;
#pragma unroll
    for (int p = 0; p < 8; p++) { acc[0][t][p] = bv; acc[1][t][p] = bv; }
  }

  int rowA0 = r0 + rl;
  int rowA1 = r0 + 16 + rl;
  if (rowA1 >= M) rowA1 = rowA0;          // clamp; results discarded by store guard
  const h16* a0p = A + (size_t)rowA0 * lda;
  const h16* a1p = A + (size_t)rowA1 * lda;

#pragma unroll
  for (int ks = 0; ks < K / 32; ks++) {
    v16h a0 = frag_load(a0p + ks * 32, hl);
    v16h a1 = frag_load(a1p + ks * 32, hl);
#pragma unroll
    for (int t = 0; t < 4; t++) {
      v16h b = frag_load(Wt + (size_t)(t * 16 + rl) * K + ks * 32, hl);
      acc[0][t] = __builtin_amdgcn_wmma_f32_16x16x32_f16(
          false, a0, false, b, (short)0, acc[0][t], false, false);
      acc[1][t] = __builtin_amdgcn_wmma_f32_16x16x32_f16(
          false, a1, false, b, (short)0, acc[1][t], false, false);
    }
  }

#pragma unroll
  for (int m = 0; m < 2; m++) {
#pragma unroll
    for (int t = 0; t < 4; t++) {
#pragma unroll
      for (int p = 0; p < 8; p++) {
        int row = r0 + m * 16 + p + (hl << 3);
        if (row < M) {
          int col = t * 16 + rl;
          size_t idx = (size_t)row * 64 + col;
          float v = acc[m][t][p];
          if (OUTH) {
            h16* C = (h16*)Cout;
            C[idx] = (h16)(ACCUM ? (float)C[idx] + v : v);
          } else {
            float* C = (float*)Cout;
            C[idx] = ACCUM ? C[idx] + v : v;
          }
        }
      }
    }
  }
}

// ---------------------------------------------------------------------------
// Dilated k=3 conv over node dim: H = relu(sum_kk shift(F,(kk-1)*d) @ Wt3[kk] + b)
// Two 16-row tiles per wave, zero-padded fragments outside [0,M).
// ---------------------------------------------------------------------------
__global__ void convd16k(const h16* __restrict__ F, const h16* __restrict__ Wt3,
                         const float* __restrict__ bias, h16* __restrict__ H,
                         int M, int dil) {
  int wave = threadIdx.x >> 5, lane = threadIdx.x & 31;
  int r0 = (blockIdx.x * 8 + wave) * 32;
  if (r0 >= M) return;
  int rl = lane & 15, hl = lane >> 4;

  v8f acc[2][4];
#pragma unroll
  for (int t = 0; t < 4; t++) {
    float bv = bias[t * 16 + rl];
#pragma unroll
    for (int p = 0; p < 8; p++) { acc[0][t][p] = bv; acc[1][t][p] = bv; }
  }

#pragma unroll
  for (int kk = 0; kk < 3; kk++) {
    int base = (kk - 1) * dil;
    int row0 = r0 + rl + base;
    int row1 = r0 + 16 + rl + base;
    bool ok0 = (row0 >= 0) && (row0 < M);
    bool ok1 = (row1 >= 0) && (row1 < M);
    const h16* a0p = F + (size_t)(ok0 ? row0 : 0) * 64;
    const h16* a1p = F + (size_t)(ok1 ? row1 : 0) * 64;
#pragma unroll
    for (int ks = 0; ks < 2; ks++) {
      v16h a0 = frag_zero();
      v16h a1 = frag_zero();
      if (ok0) a0 = frag_load(a0p + ks * 32, hl);
      if (ok1) a1 = frag_load(a1p + ks * 32, hl);
#pragma unroll
      for (int t = 0; t < 4; t++) {
        v16h b = frag_load(Wt3 + kk * 4096 + (size_t)(t * 16 + rl) * 64 + ks * 32, hl);
        acc[0][t] = __builtin_amdgcn_wmma_f32_16x16x32_f16(
            false, a0, false, b, (short)0, acc[0][t], false, false);
        acc[1][t] = __builtin_amdgcn_wmma_f32_16x16x32_f16(
            false, a1, false, b, (short)0, acc[1][t], false, false);
      }
    }
  }

#pragma unroll
  for (int m = 0; m < 2; m++) {
#pragma unroll
    for (int t = 0; t < 4; t++) {
#pragma unroll
      for (int p = 0; p < 8; p++) {
        int row = r0 + m * 16 + p + (hl << 3);
        if (row < M) {
          int col = t * 16 + rl;
          float v = acc[m][t][p];
          H[(size_t)row * 64 + col] = (h16)(v > 0.0f ? v : 0.0f);
        }
      }
    }
  }
}

// ---------------------------------------------------------------------------
// EdgeConv per-edge layer 2: h1 = relu(A[dst]+B[src]+b1); out = h1@W2 + b2;
// atomicMax(ordered) scatter at dst for masked edges. 16 edges per wave.
// ---------------------------------------------------------------------------
__global__ void edgeconvk(const float* __restrict__ Abuf, const float* __restrict__ Bbuf,
                          const long long* __restrict__ src, const long long* __restrict__ dst,
                          const long long* __restrict__ attr,
                          const float* __restrict__ b1, const h16* __restrict__ Wt2,
                          const float* __restrict__ b2,
                          unsigned* __restrict__ outU, int E, int branch) {
  __shared__ h16 hbuf[8][16][72];   // padded stride to spread LDS banks
  __shared__ int sdst[8][16];
  __shared__ int smask[8][16];

  int wave = threadIdx.x >> 5, lane = threadIdx.x & 31;
  int rl = lane & 15, hl = lane >> 4;
  int e0 = (blockIdx.x * 8 + wave) * 16;
  int e = e0 + rl;
  bool valid = e < E;
  int ec = valid ? e : (E - 1);

  long long s = src[ec], d = dst[ec];
  const float* ap = Abuf + (size_t)d * 64 + hl * 32;
  const float* bp = Bbuf + (size_t)s * 64 + hl * 32;
  h16* hrow = &hbuf[wave][rl][hl * 32];
#pragma unroll
  for (int q = 0; q < 8; q++) {
    v4f av = *(const v4f*)(ap + q * 4);
    v4f bv = *(const v4f*)(bp + q * 4);
    v4f b1v = *(const v4f*)(b1 + hl * 32 + q * 4);
#pragma unroll
    for (int j = 0; j < 4; j++) {
      float v = av[j] + bv[j] + b1v[j];
      hrow[q * 4 + j] = (h16)(v > 0.0f ? v : 0.0f);
    }
  }
  if (hl == 0) {
    sdst[wave][rl] = (int)d;
    long long a = attr[ec];
    int m = (branch == 0) ? (a <= 0) : (branch == 1) ? (a >= 0) : 1;
    smask[wave][rl] = valid ? m : 0;
  }
  __syncthreads();

  v8f acc[4];
#pragma unroll
  for (int t = 0; t < 4; t++) {
    float bv = b2[t * 16 + rl];
#pragma unroll
    for (int p = 0; p < 8; p++) acc[t][p] = bv;
  }
#pragma unroll
  for (int ks = 0; ks < 2; ks++) {
    v16h a = frag_load(&hbuf[wave][rl][ks * 32], hl);
#pragma unroll
    for (int t = 0; t < 4; t++) {
      v16h b = frag_load(Wt2 + (size_t)(t * 16 + rl) * 64 + ks * 32, hl);
      acc[t] = __builtin_amdgcn_wmma_f32_16x16x32_f16(
          false, a, false, b, (short)0, acc[t], false, false);
    }
  }

#pragma unroll
  for (int t = 0; t < 4; t++) {
#pragma unroll
    for (int p = 0; p < 8; p++) {
      int row = p + (hl << 3);
      int col = t * 16 + rl;
      if (smask[wave][row]) {
        atomicMax(&outU[(size_t)sdst[wave][row] * 64 + col], ordF(acc[t][p]));
      }
    }
  }
}

// ---------------------------------------------------------------------------
// BatchNorm statistics (sum, sumsq per channel) from f32 or ordered-uint src.
// ---------------------------------------------------------------------------
__global__ void statsk(const float* __restrict__ Xf, const unsigned* __restrict__ Xu,
                       float* __restrict__ stats, int N) {
  __shared__ float ls[4][64], ls2[4][64];
  int c = threadIdx.x & 63, y = threadIdx.x >> 6;
  float s = 0.0f, s2 = 0.0f;
  for (int n = blockIdx.x * 4 + y; n < N; n += gridDim.x * 4) {
    size_t idx = (size_t)n * 64 + c;
    float v = Xf ? Xf[idx] : decU(Xu[idx]);
    s += v; s2 += v * v;
  }
  ls[y][c] = s; ls2[y][c] = s2;
  __syncthreads();
  if (y == 0) {
    float t = ls[0][c] + ls[1][c] + ls[2][c] + ls[3][c];
    float t2 = ls2[0][c] + ls2[1][c] + ls2[2][c] + ls2[3][c];
    atomicAdd(&stats[c], t);
    atomicAdd(&stats[64 + c], t2);
  }
}

__global__ void bnfin(const float* __restrict__ stats, const float* __restrict__ gamma,
                      const float* __restrict__ beta, float* __restrict__ ss, float n) {
  int c = threadIdx.x;
  if (c >= 64) return;
  float mean = stats[c] / n;
  float var = stats[64 + c] / n - mean * mean;
  float sc = gamma[c] * rsqrtf(var + 1e-5f);
  ss[c] = sc;
  ss[64 + c] = beta[c] - mean * sc;
}

__global__ void bnapply_x(const float* __restrict__ x, const float* __restrict__ ss,
                          h16* __restrict__ xn, int total) {
  int i = blockIdx.x * blockDim.x + threadIdx.x;
  if (i >= total) return;
  int c = i & 63;
  float v = x[i] * ss[c] + ss[64 + c];
  xn[i] = (h16)(v > 0.0f ? v : 0.0f);
}

// decode edgeconv result, BN+relu, write f16 into cols 0..63 of [N,192] buffer
__global__ void bnapply_dec(const unsigned* __restrict__ U, const float* __restrict__ ss,
                            h16* __restrict__ in192, int N) {
  int i = blockIdx.x * blockDim.x + threadIdx.x;
  if (i >= N * 64) return;
  int n = i >> 6, c = i & 63;
  float v = decU(U[i]) * ss[c] + ss[64 + c];
  in192[(size_t)n * 192 + c] = (h16)(v > 0.0f ? v : 0.0f);
}

// ---------------------------------------------------------------------------
// RGCN: scatter-add of h[src] per relation + degree count (aggregate-then-GEMM)
// ---------------------------------------------------------------------------
__global__ void rgcn_scatter(const h16* __restrict__ in192, const long long* __restrict__ src,
                             const long long* __restrict__ dst, const long long* __restrict__ attr,
                             float* __restrict__ S, float* __restrict__ cnt,
                             int E, int N, int branch) {
  int i = blockIdx.x * blockDim.x + threadIdx.x;
  if (i >= 2 * E) return;
  int e = i >> 1, half = i & 1;
  long long a = attr[e];
  int m = (branch == 0) ? (a <= 0) : (branch == 1) ? (a >= 0) : 1;
  if (!m) return;
  int r = (a != -2) ? 1 : 0;
  int si = (int)src[e], di = (int)dst[e];
  const h16* hp = in192 + (size_t)si * 192 + half * 32;
  float* Sp = S + ((size_t)r * N + di) * 64 + half * 32;
#pragma unroll
  for (int c = 0; c < 32; c++) atomicAdd(&Sp[c], (float)hp[c]);
  if (half == 0) atomicAdd(&cnt[(size_t)r * N + di], 1.0f);
}

__global__ void rgcn_norm(const float* __restrict__ S, const float* __restrict__ cnt,
                          h16* __restrict__ in192, int N) {
  int i = blockIdx.x * blockDim.x + threadIdx.x;
  if (i >= N * 64) return;
  int n = i >> 6, c = i & 63;
  float c0 = fmaxf(cnt[n], 1.0f), c1 = fmaxf(cnt[(size_t)N + n], 1.0f);
  in192[(size_t)n * 192 + 64 + c]  = (h16)(S[(size_t)n * 64 + c] / c0);
  in192[(size_t)n * 192 + 128 + c] = (h16)(S[((size_t)N + n) * 64 + c] / c1);
}

// ---------------------------------------------------------------------------
// softmax over 64 channels per node -> f16
// ---------------------------------------------------------------------------
__global__ void softmax64(const float* __restrict__ X, h16* __restrict__ Y, int M) {
  int n = blockIdx.x * blockDim.x + threadIdx.x;
  if (n >= M) return;
  const float* p = X + (size_t)n * 64;
  float v[64];
  float mx = -3.4e38f;
#pragma unroll
  for (int q = 0; q < 16; q++) {
    v4f t = *(const v4f*)(p + q * 4);
#pragma unroll
    for (int j = 0; j < 4; j++) { v[q * 4 + j] = t[j]; mx = fmaxf(mx, t[j]); }
  }
  float sum = 0.0f;
#pragma unroll
  for (int i = 0; i < 64; i++) { v[i] = __expf(v[i] - mx); sum += v[i]; }
  float inv = 1.0f / sum;
  h16* o = Y + (size_t)n * 64;
#pragma unroll
  for (int i = 0; i < 64; i++) o[i] = (h16)(v[i] * inv);
}

// ---------------------------------------------------------------------------
// Weight prep kernels (f32 -> f16, transposed into Wt[n][k] layout)
// ---------------------------------------------------------------------------
__global__ void prep_w1(const float* __restrict__ W1, h16* __restrict__ Wa,
                        h16* __restrict__ Wb) {  // W1 [128][64]
  int i = blockIdx.x * blockDim.x + threadIdx.x;
  if (i >= 64 * 64) return;
  int k = i >> 6, n = i & 63;
  float top = W1[k * 64 + n], bot = W1[(64 + k) * 64 + n];
  Wa[n * 64 + k] = (h16)(top - bot);
  Wb[n * 64 + k] = (h16)bot;
}

__global__ void prep_t(const float* __restrict__ W, h16* __restrict__ Wt,
                       int K, int Nc, int ldw, int kofs) {  // W[K][Nc] -> Wt[n][kofs+k]
  int i = blockIdx.x * blockDim.x + threadIdx.x;
  if (i >= K * Nc) return;
  int k = i / Nc, n = i % Nc;
  Wt[(size_t)n * ldw + kofs + k] = (h16)W[(size_t)k * Nc + n];
}

__global__ void prep_cast(const float* __restrict__ W, h16* __restrict__ Wt, int len) {
  int i = blockIdx.x * blockDim.x + threadIdx.x;   // [co][ci] layout kept as-is
  if (i >= len) return;
  Wt[i] = (h16)W[i];
}

__global__ void prep_conv3(const float* __restrict__ w, h16* __restrict__ Wt3) {
  int i = blockIdx.x * blockDim.x + threadIdx.x;   // w [64][64][3] -> Wt3[3][64][64]
  if (i >= 64 * 64 * 3) return;
  int kk = i / 4096, rest = i % 4096;
  int co = rest >> 6, ci = rest & 63;
  Wt3[kk * 4096 + co * 64 + ci] = (h16)w[(size_t)(co * 64 + ci) * 3 + kk];
}

// ---------------------------------------------------------------------------
extern "C" void kernel_launch(void* const* d_in, const int* in_sizes, int n_in,
                              void* d_out, int out_size, void* d_ws, size_t ws_size,
                              hipStream_t stream) {
  (void)n_in; (void)out_size; (void)ws_size;
  const int N = in_sizes[0] / 64;
  const int E = in_sizes[2];

  const float*     x    = (const float*)d_in[0];
  const long long* src  = (const long long*)d_in[1];
  const long long* dst  = src + E;
  const long long* attr = (const long long*)d_in[2];
  const float*     bng  = (const float*)d_in[3];
  const float*     bnb  = (const float*)d_in[4];
  auto ecp  = [&](int b, int j) { return (const float*)d_in[5 + 6 * b + j]; };   // W1,b1,W2,b2,gamma,beta
  auto rgp  = [&](int b, int j) { return (const float*)d_in[23 + 3 * b + j]; };  // weight,root,bias
  auto refp = [&](int s, int j) { return (const float*)d_in[32 + 44 * s + j]; };

  // workspace carve (bump allocator, 256B aligned)
  size_t off = 0;
  auto walloc = [&](size_t bytes) -> void* {
    void* p = (char*)d_ws + off;
    off += (bytes + 255) & ~(size_t)255;
    return p;
  };
  h16*      xn    = (h16*)walloc((size_t)N * 64 * 2);
  float*    Abuf  = (float*)walloc((size_t)N * 64 * 4);
  float*    Bbuf  = (float*)walloc((size_t)N * 64 * 4);
  unsigned* outU  = (unsigned*)walloc((size_t)N * 64 * 4);
  h16*      in192 = (h16*)walloc((size_t)N * 192 * 2);
  float*    S     = (float*)walloc((size_t)2 * N * 64 * 4);
  float*    cnt   = (float*)walloc((size_t)2 * N * 4);
  float*    out3  = (float*)walloc((size_t)N * 64 * 4);
  float*    stats = (float*)walloc(128 * 4);
  float*    ss    = (float*)walloc(128 * 4);
  h16* Wa     = (h16*)walloc(4096 * 2);
  h16* Wb     = (h16*)walloc(4096 * 2);
  h16* Wt2    = (h16*)walloc(4096 * 2);
  h16* Wt192  = (h16*)walloc((size_t)64 * 192 * 2);
  h16* Wtin   = (h16*)walloc(4096 * 2);
  h16* Wtout  = (h16*)walloc(4096 * 2);
  h16* Wt3all = (h16*)walloc((size_t)10 * 3 * 4096 * 2);
  h16* Wt1all = (h16*)walloc((size_t)10 * 4096 * 2);
  // refinement buffers reuse graph-phase scratch
  h16* smb = xn;
  h16* F   = (h16*)Abuf;
  h16* H   = (h16*)Bbuf;

  const int gb = (N + 255) / 256;          // 8 waves/block, 32 rows/wave

  // input BN + relu -> xn (f16)
  hipMemsetAsync(stats, 0, 128 * 4, stream);
  statsk<<<240, 256, 0, stream>>>(x, nullptr, stats, N);
  bnfin<<<1, 64, 0, stream>>>(stats, bng, bnb, ss, (float)N);
  bnapply_x<<<(N * 64 + 255) / 256, 256, 0, stream>>>(x, ss, xn, N * 64);

  hipMemsetAsync(out3, 0, (size_t)N * 64 * 4, stream);

  for (int b = 0; b < 3; b++) {
    // weight prep
    prep_w1<<<16, 256, 0, stream>>>(ecp(b, 0), Wa, Wb);
    prep_t<<<16, 256, 0, stream>>>(ecp(b, 2), Wt2, 64, 64, 64, 0);
    prep_t<<<16, 256, 0, stream>>>(rgp(b, 1), Wt192, 64, 64, 192, 0);          // root
    prep_t<<<16, 256, 0, stream>>>(rgp(b, 0), Wt192, 64, 64, 192, 64);         // W_r=0
    prep_t<<<16, 256, 0, stream>>>(rgp(b, 0) + 4096, Wt192, 64, 64, 192, 128); // W_r=1
    // EdgeConv factored layer-1 node GEMMs
    gemm16<64, false, false><<<gb, 256, 0, stream>>>(xn, 64, Wa, nullptr, Abuf, N);
    gemm16<64, false, false><<<gb, 256, 0, stream>>>(xn, 64, Wb, nullptr, Bbuf, N);
    // per-edge layer-2 + max-aggregation
    hipMemsetAsync(outU, 0, (size_t)N * 64 * 4, stream);
    edgeconvk<<<(E + 127) / 128, 256, 0, stream>>>(Abuf, Bbuf, src, dst, attr,
                                                   ecp(b, 1), Wt2, ecp(b, 3), outU, E, b);
    // BN over edgeconv output, relu, -> in192 cols 0..63
    hipMemsetAsync(stats, 0, 128 * 4, stream);
    statsk<<<240, 256, 0, stream>>>(nullptr, outU, stats, N);
    bnfin<<<1, 64, 0, stream>>>(stats, ecp(b, 4), ecp(b, 5), ss, (float)N);
    bnapply_dec<<<(N * 64 + 255) / 256, 256, 0, stream>>>(outU, ss, in192, N);
    // RGCN aggregate-then-GEMM
    hipMemsetAsync(S, 0, (size_t)2 * N * 64 * 4, stream);
    hipMemsetAsync(cnt, 0, (size_t)2 * N * 4, stream);
    rgcn_scatter<<<(2 * E + 255) / 256, 256, 0, stream>>>(in192, src, dst, attr, S, cnt, E, N, b);
    rgcn_norm<<<(N * 64 + 255) / 256, 256, 0, stream>>>(S, cnt, in192, N);
    gemm16<192, true, false><<<gb, 256, 0, stream>>>(in192, 192, Wt192, rgp(b, 2), out3, N);
  }

  // xr0 -> d_out slot 0 ([4,N,64] node-major matches our layout)
  hipMemcpyAsync(d_out, out3, (size_t)N * 64 * 4, hipMemcpyDeviceToDevice, stream);

  for (int s = 0; s < 3; s++) {
    prep_cast<<<16, 256, 0, stream>>>(refp(s, 0), Wtin, 4096);
    for (int i = 0; i < 10; i++) {
      prep_conv3<<<48, 256, 0, stream>>>(refp(s, 2 + 4 * i), Wt3all + i * 3 * 4096);
      prep_cast<<<16, 256, 0, stream>>>(refp(s, 4 + 4 * i), Wt1all + i * 4096, 4096);
    }
    prep_cast<<<16, 256, 0, stream>>>(refp(s, 42), Wtout, 4096);

    const float* xin  = (const float*)d_out + (size_t)s * N * 64;
    float*       xout = (float*)d_out + (size_t)(s + 1) * N * 64;

    softmax64<<<(N + 255) / 256, 256, 0, stream>>>(xin, smb, N);
    gemm16<64, false, true><<<gb, 256, 0, stream>>>(smb, 64, Wtin, refp(s, 1), F, N);
    for (int i = 0; i < 10; i++) {
      convd16k<<<gb, 256, 0, stream>>>(F, Wt3all + i * 3 * 4096, refp(s, 3 + 4 * i), H, N, 1 << i);
      gemm16<64, true, true><<<gb, 256, 0, stream>>>(H, 64, Wt1all + i * 4096, refp(s, 5 + 4 * i), F, N);
    }
    gemm16<64, false, false><<<gb, 256, 0, stream>>>(F, 64, Wtout, refp(s, 43), xout, N);
  }
}
